// SkyLadder_22256520527988
// MI455X (gfx1250) — matrix operand order repaired and checked by reading, not compiled, not run on hardware
//
#include <hip/hip_runtime.h>
#include <math.h>

// Problem constants (match reference)
#define Bc 2
#define Tc 2048
#define Vc 32000
#define Dc 1024
#define Lc 4
#define Hc 16
#define HDc 64

typedef __attribute__((ext_vector_type(16))) __bf16 v16bf;
typedef __attribute__((ext_vector_type(8)))  __bf16 v8bf;
typedef __attribute__((ext_vector_type(8)))  float  v8f;
typedef __attribute__((ext_vector_type(4)))  unsigned int u32x4;
typedef __attribute__((ext_vector_type(8)))  int i32x8;
typedef __attribute__((ext_vector_type(4)))  int i32x4;

static __device__ __forceinline__ v8bf ldb8(const __bf16* p) {
    return *reinterpret_cast<const v8bf*>(p);
}
static __device__ __forceinline__ v16bf mk16(v8bf lo, v8bf hi) {
    v16bf r;
#pragma unroll
    for (int i = 0; i < 8; i++) { r[i] = lo[i]; r[8 + i] = hi[i]; }
    return r;
}

// ---------------------------------------------------------------------------
// Tensor Data Mover: async 2-D tile load (rows x 32 bf16, row-major, packed)
// D# built per CDNA5 ISA ch.8: group0 {count=1, lds_addr, global_addr, type=2},
// group1 {data_size=2B, tensor_dim0=K, tensor_dim1=rowsTotal, tile=32 x rows,
// dim0_stride=K}. Tracked by TENSORcnt.
// clang-23 (amdgpu-toolchain) takes 6 args (extra trailing int32x8);
// clang-22 (ROCm 7.2) takes 5.
// ---------------------------------------------------------------------------
#if __has_builtin(__builtin_amdgcn_tensor_load_to_lds)
#define USE_TDM 1
static __device__ __forceinline__ void tdm_load_2d(const __bf16* gptr,
                                                   unsigned int lds_byte_off,
                                                   int tile_rows, int tensor_rows,
                                                   int K) {
    unsigned long long ga = (unsigned long long)(uintptr_t)gptr;
    u32x4 g0;
    g0[0] = 1u;                                              // count=1
    g0[1] = lds_byte_off;                                    // lds_addr
    g0[2] = (unsigned int)(ga & 0xFFFFFFFFu);                // global_addr lo
    g0[3] = (unsigned int)((ga >> 32) & 0x01FFFFFFu) | (2u << 30);  // + type=2
    unsigned int td0 = (unsigned int)K;                      // tensor_dim0
    unsigned int td1 = (unsigned int)tensor_rows;            // tensor_dim1
    unsigned int st0 = (unsigned int)K;                      // dim0 stride
    i32x8 g1;
    g1[0] = (int)(1u << 16);                                 // data_size = 2 bytes
    g1[1] = (int)((td0 & 0xFFFFu) << 16);                    // atomic_addr=0 | td0 lo
    g1[2] = (int)((td0 >> 16) | ((td1 & 0xFFFFu) << 16));
    g1[3] = (int)((td1 >> 16) | (32u << 16));                // tile_dim0 = 32
    g1[4] = (int)((unsigned int)tile_rows & 0xFFFFu);        // tile_dim1, tile_dim2=0
    g1[5] = (int)st0;                                        // stride lo32
    g1[6] = 0;                                               // stride hi16, dim1_stride=0
    g1[7] = 0;
    i32x4 gz4 = {0, 0, 0, 0};
#if defined(__clang_major__) && (__clang_major__ >= 23)
    i32x8 gz8 = {0, 0, 0, 0, 0, 0, 0, 0};
    __builtin_amdgcn_tensor_load_to_lds(g0, g1, gz4, gz4, gz8, 0);
#else
    __builtin_amdgcn_tensor_load_to_lds(g0, g1, gz4, gz4, 0);
#endif
}
#else
#define USE_TDM 0
#endif

// ---------------------------------------------------------------------------
// fp32 -> bf16 convert (plain) and transpose-convert (W[R,C] -> Wt[C,R])
// ---------------------------------------------------------------------------
__global__ void conv_bf16(const float* __restrict__ s, __bf16* __restrict__ d, size_t n) {
    size_t i = (size_t)blockIdx.x * 256 + threadIdx.x;
    if (i < n) d[i] = (__bf16)s[i];
}

__global__ void tconv_bf16(const float* __restrict__ s, __bf16* __restrict__ d, int R, int C) {
    size_t i = (size_t)blockIdx.x * 256 + threadIdx.x;
    size_t n = (size_t)R * C;
    if (i < n) {
        int r = (int)(i / C);
        int c = (int)(i % C);
        d[(size_t)c * R + r] = (__bf16)s[i];
    }
}

// ---------------------------------------------------------------------------
// token + position embedding: x[b,t,:] = wte[id] + wpe[t]
// ---------------------------------------------------------------------------
__global__ void embed_kernel(const int* __restrict__ ids, const float* __restrict__ wte,
                             const float* __restrict__ wpe, float* __restrict__ x) {
    int t  = blockIdx.x;            // 0 .. B*T-1
    int tok = ids[t];
    int tt  = t % Tc;
    for (int i = threadIdx.x; i < Dc; i += 256)
        x[(size_t)t * Dc + i] = wte[(size_t)tok * Dc + i] + wpe[(size_t)tt * Dc + i];
}

// ---------------------------------------------------------------------------
// LayerNorm (one wave per token), bf16 output for WMMA A operand
// ---------------------------------------------------------------------------
__global__ __launch_bounds__(256) void ln_bf16(const float* __restrict__ x,
                                               const float* __restrict__ g,
                                               const float* __restrict__ bb,
                                               __bf16* __restrict__ out) {
    const int lane = threadIdx.x & 31;
    const int wave = threadIdx.x >> 5;
    const int tok  = blockIdx.x * 8 + wave;
    const float* xr = x + (size_t)tok * Dc;
    float s = 0.f, s2 = 0.f;
    for (int i = lane; i < Dc; i += 32) { float v = xr[i]; s += v; s2 += v * v; }
#pragma unroll
    for (int off = 1; off < 32; off <<= 1) {
        s  += __shfl_xor(s,  off, 32);
        s2 += __shfl_xor(s2, off, 32);
    }
    const float mean = s * (1.f / Dc);
    const float var  = s2 * (1.f / Dc) - mean * mean;
    const float inv  = rsqrtf(var + 1e-5f);
    for (int i = lane; i < Dc; i += 32)
        out[(size_t)tok * Dc + i] = (__bf16)((xr[i] - mean) * inv * g[i] + bb[i]);
}

// ---------------------------------------------------------------------------
// Per-head V transpose: vt[(b*H+h)*64+d][t] = qkv[b,t, 2D + h*64 + d]
// ---------------------------------------------------------------------------
__global__ void vtrans_kernel(const __bf16* __restrict__ qkv, __bf16* __restrict__ vt) {
    size_t i = (size_t)blockIdx.x * 256 + threadIdx.x;
    const size_t n = (size_t)Bc * Tc * Dc;
    if (i >= n) return;
    int dfull = (int)(i % Dc);
    int t     = (int)((i / Dc) % Tc);
    int b     = (int)(i / ((size_t)Dc * Tc));
    int h = dfull >> 6, d = dfull & 63;
    vt[((size_t)(b * Hc + h) * HDc + d) * Tc + t] =
        qkv[(size_t)(b * Tc + t) * (3 * Dc) + 2 * Dc + dfull];
}

// ---------------------------------------------------------------------------
// GEMM: C[M,N] = A[M,K](bf16) @ Bt[N,K](bf16)^T with epilogue modes
//   mode 0: outf = acc (+bias)        mode 1: res += acc + bias
//   mode 2: outb = bf16(acc+bias)     mode 3: outb = bf16(gelu(acc+bias))
// Block 256 thr = 8 waves (2M x 4N), block tile 128x256, wave tile 64x64
// (4x4 v_wmma_f32_16x16x32_bf16 accumulators). A/B K-step tiles staged in
// LDS by the Tensor Data Mover (double-buffered, TENSORcnt) when available.
// All dims: M %128==0, N %256==0, K %32==0.
// ---------------------------------------------------------------------------
__global__ __launch_bounds__(256) void gemm_bf16_wmma(
    const __bf16* __restrict__ A, const __bf16* __restrict__ Bt,
    const float* __restrict__ bias,
    float* __restrict__ outf, __bf16* __restrict__ outb, float* __restrict__ res,
    int M, int N, int K, int mode) {
    __shared__ alignas(16) __bf16 ldsA[2][128 * 32];
    __shared__ alignas(16) __bf16 ldsB[2][256 * 32];

    const int tid  = threadIdx.x;
    const int lane = tid & 31;
    const int wave = tid >> 5;
    const int l16  = lane & 15;
    const int half = lane >> 4;
    const int wm   = wave >> 2;          // 0..1
    const int wn   = wave & 3;           // 0..3
    const int mBlk = blockIdx.y * 128;
    const int nBlk = blockIdx.x * 256;

    v8f zero = {};
    v8f acc[4][4];
#pragma unroll
    for (int i = 0; i < 4; i++)
#pragma unroll
        for (int j = 0; j < 4; j++) acc[i][j] = zero;

    const int kaA = half ? 8 : 0;
    const int kaB = half ? 8 : 0;
    const int nk  = K / 32;

#if USE_TDM
    if (wave == 0) {
        tdm_load_2d(A  + (size_t)mBlk * K, (unsigned int)(uintptr_t)&ldsA[0][0], 128, M, K);
        tdm_load_2d(Bt + (size_t)nBlk * K, (unsigned int)(uintptr_t)&ldsB[0][0], 256, N, K);
    }
#endif

    for (int ik = 0; ik < nk; ik++) {
        const int buf = ik & 1;
#if USE_TDM
        if (wave == 0) __builtin_amdgcn_s_wait_tensorcnt(0);
        __syncthreads();                         // tile `buf` ready for everyone
        if (wave == 0 && ik + 1 < nk) {          // prefetch next tile (other buffer)
            const int kk2 = (ik + 1) * 32;
            tdm_load_2d(A  + (size_t)mBlk * K + kk2,
                        (unsigned int)(uintptr_t)&ldsA[buf ^ 1][0], 128, M, K);
            tdm_load_2d(Bt + (size_t)nBlk * K + kk2,
                        (unsigned int)(uintptr_t)&ldsB[buf ^ 1][0], 256, N, K);
        }
#else
        const int kk = ik * 32;
        __syncthreads();                         // WAR: previous compute done
        // cooperative staging: one 8-bf16 (16B) copy per chunk, chunks at cols 0/16
        for (int c = tid; c < 128 * 2; c += 256) {
            int r = c >> 1, col = (c & 1) * 16;
            *reinterpret_cast<v8bf*>(&ldsA[buf][r * 32 + col]) =
                ldb8(A + (size_t)(mBlk + r) * K + kk + col);
        }
        for (int c = tid; c < 256 * 2; c += 256) {
            int r = c >> 1, col = (c & 1) * 16;
            *reinterpret_cast<v8bf*>(&ldsB[buf][r * 32 + col]) =
                ldb8(Bt + (size_t)(nBlk + r) * K + kk + col);
        }
        __syncthreads();                         // tile `buf` ready
#endif
        const __bf16* lA = &ldsA[buf][0];
        const __bf16* lB = &ldsB[buf][0];
        v16bf afr[4];
#pragma unroll
        for (int mi = 0; mi < 4; mi++) {
            const __bf16* ap = lA + (wm * 64 + mi * 16 + l16) * 32 + kaA;
            afr[mi] = mk16(ldb8(ap), ldb8(ap + 16));
        }
        v16bf bfr[4];
#pragma unroll
        for (int ni = 0; ni < 4; ni++) {
            const __bf16* bp = lB + (wn * 64 + ni * 16 + l16) * 32 + kaB;
            bfr[ni] = mk16(ldb8(bp), ldb8(bp + 16));
        }
#pragma unroll
        for (int mi = 0; mi < 4; mi++)
#pragma unroll
            for (int ni = 0; ni < 4; ni++)
                acc[mi][ni] = __builtin_amdgcn_wmma_f32_16x16x32_bf16(
                    false, afr[mi], false, bfr[ni], (short)0, acc[mi][ni], false, false);
    }

#pragma unroll
    for (int ni = 0; ni < 4; ni++) {
        const int col = nBlk + wn * 64 + ni * 16 + l16;
        const float bv = bias ? bias[col] : 0.f;
#pragma unroll
        for (int mi = 0; mi < 4; mi++) {
#pragma unroll
            for (int j = 0; j < 8; j++) {
                const int row = mBlk + wm * 64 + mi * 16 + half * 8 + j;
                float v = acc[mi][ni][j] + bv;
                const size_t idx = (size_t)row * N + col;
                if (mode == 0) {
                    outf[idx] = v;
                } else if (mode == 1) {
                    res[idx] += v;
                } else if (mode == 2) {
                    outb[idx] = (__bf16)v;
                } else {
                    float t = 0.7978845608028654f * (v + 0.044715f * v * v * v);
                    outb[idx] = (__bf16)(0.5f * v * (1.f + tanhf(t)));
                }
            }
        }
    }
}

// ---------------------------------------------------------------------------
// SkyLadder masked flash attention. One wave per 16-query tile.
// S = Q K^T via WMMA (contraction over d=64), online softmax, P V via WMMA
// (contraction over 32 keys), P transposed C->A layout through LDS.
// ---------------------------------------------------------------------------
static __device__ __forceinline__ int mask_len(int it) {
    const int cp = Tc * 8;
    if (it >= cp) return Tc;
    float progress = (float)it / (float)cp;
    return 512 + (int)((float)(Tc - 512) * progress);
}

__global__ __launch_bounds__(256) void attn_wmma(
    const __bf16* __restrict__ qkv, const __bf16* __restrict__ vT,
    __bf16* __restrict__ outb, const int* __restrict__ iter_num) {
    __shared__ alignas(16) __bf16 pbuf[8][16 * 32];
    const int lane = threadIdx.x & 31;
    const int wave = threadIdx.x >> 5;
    const int l16  = lane & 15;
    const int half = lane >> 4;
    const int nqb  = Tc / 128;
    const int bh   = blockIdx.x / nqb;
    const int b    = bh / Hc;
    const int h    = bh % Hc;
    const int q0   = (blockIdx.x % nqb) * 128 + wave * 16;

    const int m = mask_len(iter_num[0]);

    // Q fragments (A layout), two K-chunks over head dim 64
    v16bf qa[2];
#pragma unroll
    for (int c = 0; c < 2; c++) {
        const __bf16* qp = qkv + (size_t)(b * Tc + q0 + l16) * (3 * Dc)
                         + h * HDc + c * 32 + (half ? 8 : 0);
        qa[c] = mk16(ldb8(qp), ldb8(qp + 16));
    }

    v8f zero = {};
    v8f o[4];
#pragma unroll
    for (int t = 0; t < 4; t++) o[t] = zero;
    float rmax[8], rsum[8];
#pragma unroll
    for (int j = 0; j < 8; j++) { rmax[j] = -1e30f; rsum[j] = 0.f; }

    const int kstart = (q0 / m) * m;          // fragment start of first row
    const int kend   = q0 + 16;               // last key + 1 (causal)
    __bf16* pb = pbuf[wave];

    for (int kc = (kstart / 32) * 32; kc < kend; kc += 32) {
        // --- scores S[16x32] ---
        v8f s[2] = {zero, zero};
#pragma unroll
        for (int ni = 0; ni < 2; ni++) {
#pragma unroll
            for (int c = 0; c < 2; c++) {
                const __bf16* kp = qkv + (size_t)(b * Tc + kc + ni * 16 + l16) * (3 * Dc)
                                 + Dc + h * HDc + c * 32 + half * 8;
                v16bf kb = mk16(ldb8(kp), ldb8(kp + 16));
                s[ni] = __builtin_amdgcn_wmma_f32_16x16x32_bf16(
                    false, qa[c], false, kb, (short)0, s[ni], false, false);
            }
        }
        // --- scale + SkyLadder mask ---
        float pv[2][8], mloc[8];
#pragma unroll
        for (int j = 0; j < 8; j++) {
            const int row = q0 + half * 8 + j;
#pragma unroll
            for (int ni = 0; ni < 2; ni++) {
                const int col = kc + ni * 16 + l16;
                float v = s[ni][j] * 0.125f;
                bool ok = (col <= row) && ((col / m) == (row / m));
                pv[ni][j] = ok ? v : -1e30f;
            }
            mloc[j] = fmaxf(pv[0][j], pv[1][j]);
        }
#pragma unroll
        for (int j = 0; j < 8; j++)
#pragma unroll
            for (int off = 1; off < 16; off <<= 1)
                mloc[j] = fmaxf(mloc[j], __shfl_xor(mloc[j], off, 32));
        // --- online softmax update ---
        float corr[8];
#pragma unroll
        for (int j = 0; j < 8; j++) {
            float nm = fmaxf(rmax[j], mloc[j]);
            corr[j] = __expf(rmax[j] - nm);
            rmax[j] = nm;
            float p0 = (pv[0][j] > -9e29f) ? __expf(pv[0][j] - nm) : 0.f;
            float p1 = (pv[1][j] > -9e29f) ? __expf(pv[1][j] - nm) : 0.f;
            pv[0][j] = p0; pv[1][j] = p1;
            float ps = p0 + p1;
#pragma unroll
            for (int off = 1; off < 16; off <<= 1) ps += __shfl_xor(ps, off, 32);
            rsum[j] = rsum[j] * corr[j] + ps;
        }
#pragma unroll
        for (int t = 0; t < 4; t++)
#pragma unroll
            for (int j = 0; j < 8; j++) o[t][j] *= corr[j];
        // --- P: C layout -> LDS -> A layout ---
#pragma unroll
        for (int ni = 0; ni < 2; ni++)
#pragma unroll
            for (int j = 0; j < 8; j++)
                pb[(half * 8 + j) * 32 + ni * 16 + l16] = (__bf16)pv[ni][j];
        __builtin_amdgcn_wave_barrier();
        asm volatile("s_wait_dscnt 0" ::: "memory");
        v16bf pa = mk16(ldb8(pb + l16 * 32 + (half ? 8 : 0)),
                        ldb8(pb + l16 * 32 + (half ? 8 : 0) + 16));
        __builtin_amdgcn_wave_barrier();
        // --- O += P @ V (contraction over 32 keys) ---
#pragma unroll
        for (int t = 0; t < 4; t++) {
            const __bf16* vp = vT + ((size_t)bh * HDc + t * 16 + l16) * Tc + kc + half * 8;
            v16bf vb = mk16(ldb8(vp), ldb8(vp + 16));
            o[t] = __builtin_amdgcn_wmma_f32_16x16x32_bf16(
                false, pa, false, vb, (short)0, o[t], false, false);
        }
    }
    // --- normalize + store bf16 ---
#pragma unroll
    for (int j = 0; j < 8; j++) {
        const int row = q0 + half * 8 + j;
        const float inv = 1.f / rsum[j];
#pragma unroll
        for (int t = 0; t < 4; t++) {
            const int d = t * 16 + l16;
            outb[(size_t)(b * Tc + row) * Dc + h * HDc + d] = (__bf16)(o[t][j] * inv);
        }
    }
}

// ---------------------------------------------------------------------------
// Host orchestration
// ---------------------------------------------------------------------------
extern "C" void kernel_launch(void* const* d_in, const int* in_sizes, int n_in,
                              void* d_out, int out_size, void* d_ws, size_t ws_size,
                              hipStream_t stream) {
    const int*   ids    = (const int*)d_in[0];
    const int*   iter   = (const int*)d_in[1];
    const float* wte    = (const float*)d_in[2];
    const float* wpe    = (const float*)d_in[3];
    const float* ln1_g  = (const float*)d_in[4];
    const float* ln1_b  = (const float*)d_in[5];
    const float* qkv_w  = (const float*)d_in[6];
    const float* qkv_b  = (const float*)d_in[7];
    const float* proj_w = (const float*)d_in[8];
    const float* proj_b = (const float*)d_in[9];
    const float* ln2_g  = (const float*)d_in[10];
    const float* ln2_b  = (const float*)d_in[11];
    const float* fc_w   = (const float*)d_in[12];
    const float* fc_b   = (const float*)d_in[13];
    const float* fc2_w  = (const float*)d_in[14];
    const float* fc2_b  = (const float*)d_in[15];
    const float* lnf_g  = (const float*)d_in[16];
    const float* lnf_b  = (const float*)d_in[17];

    char* ws = (char*)d_ws;
    size_t off = 0;
    auto alloc = [&](size_t bytes) -> char* {
        char* p = ws + off;
        off = (off + bytes + 255) & ~(size_t)255;
        return p;
    };
    const int M = Bc * Tc;  // 4096 rows
    float*  xb    = (float*) alloc((size_t)M * Dc * 4);
    __bf16* hb    = (__bf16*)alloc((size_t)M * Dc * 2);
    __bf16* qkvb  = (__bf16*)alloc((size_t)M * 3 * Dc * 2);
    __bf16* attnb = (__bf16*)alloc((size_t)M * Dc * 2);
    __bf16* actb  = (__bf16*)alloc((size_t)M * 4 * Dc * 2);
    __bf16* vtb   = (__bf16*)alloc((size_t)Bc * Hc * HDc * Tc * 2);
    __bf16* wteb  = (__bf16*)alloc((size_t)Vc * Dc * 2);
    __bf16* qkvT  = (__bf16*)alloc((size_t)Lc * 3 * Dc * Dc * 2);
    __bf16* projT = (__bf16*)alloc((size_t)Lc * Dc * Dc * 2);
    __bf16* fcT   = (__bf16*)alloc((size_t)Lc * 4 * Dc * Dc * 2);
    __bf16* fc2T  = (__bf16*)alloc((size_t)Lc * 4 * Dc * Dc * 2);

    // Weight conversion / transposition to bf16 [N,K] layout
    {
        size_t n = (size_t)Vc * Dc;
        conv_bf16<<<(unsigned)((n + 255) / 256), 256, 0, stream>>>(wte, wteb, n);
    }
    for (int l = 0; l < Lc; l++) {
        size_t n;
        n = (size_t)Dc * 3 * Dc;
        tconv_bf16<<<(unsigned)((n + 255) / 256), 256, 0, stream>>>(
            qkv_w + (size_t)l * n, qkvT + (size_t)l * n, Dc, 3 * Dc);
        n = (size_t)Dc * Dc;
        tconv_bf16<<<(unsigned)((n + 255) / 256), 256, 0, stream>>>(
            proj_w + (size_t)l * n, projT + (size_t)l * n, Dc, Dc);
        n = (size_t)Dc * 4 * Dc;
        tconv_bf16<<<(unsigned)((n + 255) / 256), 256, 0, stream>>>(
            fc_w + (size_t)l * n, fcT + (size_t)l * n, Dc, 4 * Dc);
        tconv_bf16<<<(unsigned)((n + 255) / 256), 256, 0, stream>>>(
            fc2_w + (size_t)l * n, fc2T + (size_t)l * n, 4 * Dc, Dc);
    }

    embed_kernel<<<M, 256, 0, stream>>>(ids, wte, wpe, xb);

    for (int l = 0; l < Lc; l++) {
        ln_bf16<<<M / 8, 256, 0, stream>>>(xb, ln1_g + l * Dc, ln1_b + l * Dc, hb);
        gemm_bf16_wmma<<<dim3(3 * Dc / 256, M / 128), 256, 0, stream>>>(
            hb, qkvT + (size_t)l * 3 * Dc * Dc, qkv_b + l * 3 * Dc,
            nullptr, qkvb, nullptr, M, 3 * Dc, Dc, 2);
        {
            size_t n = (size_t)M * Dc;
            vtrans_kernel<<<(unsigned)((n + 255) / 256), 256, 0, stream>>>(qkvb, vtb);
        }
        attn_wmma<<<Bc * Hc * (Tc / 128), 256, 0, stream>>>(qkvb, vtb, attnb, iter);
        gemm_bf16_wmma<<<dim3(Dc / 256, M / 128), 256, 0, stream>>>(
            attnb, projT + (size_t)l * Dc * Dc, proj_b + l * Dc,
            nullptr, nullptr, xb, M, Dc, Dc, 1);
        ln_bf16<<<M / 8, 256, 0, stream>>>(xb, ln2_g + l * Dc, ln2_b + l * Dc, hb);
        gemm_bf16_wmma<<<dim3(4 * Dc / 256, M / 128), 256, 0, stream>>>(
            hb, fcT + (size_t)l * 4 * Dc * Dc, fc_b + l * 4 * Dc,
            nullptr, actb, nullptr, M, 4 * Dc, Dc, 3);
        gemm_bf16_wmma<<<dim3(Dc / 256, M / 128), 256, 0, stream>>>(
            actb, fc2T + (size_t)l * 4 * Dc * Dc, fc2_b + l * Dc,
            nullptr, nullptr, xb, M, Dc, 4 * Dc, 1);
    }
    ln_bf16<<<M / 8, 256, 0, stream>>>(xb, lnf_g, lnf_b, hb);
    // Weight-tied LM head: logits[4096, 32000] = h @ wte^T
    gemm_bf16_wmma<<<dim3(Vc / 256, M / 128), 256, 0, stream>>>(
        hb, wteb, nullptr, (float*)d_out, nullptr, nullptr, M, Vc, Dc, 0);
}
